// MaintenanceGNN_15908558864922
// MI455X (gfx1250) — matrix operand order repaired and checked by reading, not compiled, not run on hardware
//
#include <hip/hip_runtime.h>

// ---------------- problem constants ----------------
#define N_NODES 50000
#define N_EDGES 800000
#define ESL     (N_EDGES + N_NODES)   // edges + self loops
#define D_INP   768
#define HID     256
#define HEADS   8
#define NEC     224
#define NRC     6

typedef __attribute__((ext_vector_type(16))) _Float16 v16h;
typedef __attribute__((ext_vector_type(8)))  _Float16 v8h;
typedef __attribute__((ext_vector_type(8)))  float    v8f;

// ---------------- WMMA helpers ----------------
__device__ __forceinline__ v8f v8f_zero() {
  v8f z = {0.f,0.f,0.f,0.f,0.f,0.f,0.f,0.f};
  return z;
}

__device__ __forceinline__ v8f wmma_f16(v16h a, v16h b, v8f c) {
  // D = A(16x32 f16) * B(32x16 f16) + C(16x16 f32)
  return __builtin_amdgcn_wmma_f32_16x16x32_f16(
      /*neg_a=*/false, a, /*neg_b=*/false, b,
      /*c_mod=*/(short)0, c, /*reuse_a=*/false, /*reuse_b=*/false);
}

// Async global->LDS copy (CDNA5): GV mode, LDS addr = low 32 bits of generic
// shared pointer (ISA 10.2: LDS aperture address truncates to addr[31:0]).
__device__ __forceinline__ void async_load_b128(const void* gaddr, void* lds) {
  unsigned loff = (unsigned)(uintptr_t)lds;
  asm volatile("global_load_async_to_lds_b128 %0, %1, off"
               :: "v"(loff), "v"(gaddr)
               : "memory");
}
__device__ __forceinline__ void wait_async0() {
  asm volatile("s_wait_asynccnt 0x0" ::: "memory");
}

// A fragment (16x32, f16). Per ISA layout: lane&15 = row M,
// lane>>4 selects K-halves {0..7,16..23} vs {8..15,24..31}.
// Caller passes p = &A[row*lda + k0 + (lane>>4)*8].
__device__ __forceinline__ v16h load_afrag_f32(const float* __restrict__ p) {
  v16h a;
#pragma unroll
  for (int j = 0; j < 8; ++j) {
    int kb = ((j & 4) << 2) + ((j & 3) << 1);   // j<4: 2j ; j>=4: 16+2(j-4)
    a[2*j]   = (_Float16)p[kb];
    a[2*j+1] = (_Float16)p[kb+1];
  }
  return a;
}
__device__ __forceinline__ v16h load_afrag_f16(const _Float16* __restrict__ p) {
  v16h a;
#pragma unroll
  for (int j = 0; j < 8; ++j) {
    int kb = ((j & 4) << 2) + ((j & 3) << 1);
    a[2*j]   = p[kb];
    a[2*j+1] = p[kb+1];
  }
  return a;
}
__device__ __forceinline__ v16h load_bfrag_f16(const _Float16* __restrict__ p, int ld) {
  v16h b;
#pragma unroll
  for (int s = 0; s < 16; ++s) {
    int kb = ((s & 8) << 1) + (s & 7);
    b[s] = p[(size_t)kb * ld];
  }
  return b;
}

// ---------------- utility kernels ----------------
__global__ void cvt_kernel(const float* __restrict__ in, _Float16* __restrict__ out, long long n) {
  long long i = (long long)blockIdx.x * blockDim.x + threadIdx.x;
  if (i < n) out[i] = (_Float16)in[i];
}

__global__ void cvt_pad_kernel(const float* __restrict__ in, _Float16* __restrict__ out,
                               int rows, int cols, int ldo) {
  int i = blockIdx.x * blockDim.x + threadIdx.x;
  if (i >= rows * ldo) return;
  int r = i / ldo, c = i % ldo;
  out[i] = (c < cols) ? (_Float16)in[r * cols + c] : (_Float16)0.f;
}

__global__ void fill_kernel(unsigned* __restrict__ p, unsigned v, long long n) {
  long long i = (long long)blockIdx.x * blockDim.x + threadIdx.x;
  if (i < n) p[i] = v;
}

// ---------------- generic WMMA GEMM ----------------
// C[M x Nc] = act(A[M x K](f32) @ Bp[K x Ncp](f16, col-padded) + bias).
// block = 128 threads (4 waves); wave w owns rows [bx*64 + w*16, +16);
// grid.y picks a 64-column stripe of the padded B. All WMMAs unconditional;
// only the store is column-guarded (after the last WMMA, EXEC all-1s inside loop).
__global__ void __launch_bounds__(128)
wmma_gemm_kernel(const float* __restrict__ A, const _Float16* __restrict__ Bp,
                 const float* __restrict__ bias, float* __restrict__ C,
                 int M, int K, int Nc, int Ncp, int relu) {
  int wave = __builtin_amdgcn_readfirstlane((int)(threadIdx.x >> 5));  // scalar
  int lane = threadIdx.x & 31;
  int m0 = blockIdx.x * 64 + wave * 16;
  if (m0 >= M) return;                        // scalar branch (M tail)
  int c0base = blockIdx.y * 64;
  int lrow = lane & 15, hi = lane >> 4;

  const float* arow = A + (size_t)(m0 + lrow) * K + hi * 8;
  const _Float16* brow = Bp + (size_t)(hi * 8) * Ncp + c0base + lrow;

  v8f acc[4];
#pragma unroll
  for (int t = 0; t < 4; ++t) acc[t] = v8f_zero();

  for (int k0 = 0; k0 < K; k0 += 32) {
    __builtin_prefetch(arow + k0 + 64, 0, 0);            // global_prefetch_b8
    v16h a = load_afrag_f32(arow + k0);
    const _Float16* bk = brow + (size_t)k0 * Ncp;
#pragma unroll
    for (int t = 0; t < 4; ++t) {
      v16h b = load_bfrag_f16(bk + t * 16, Ncp);
      acc[t] = wmma_f16(a, b, acc[t]);
    }
  }
#pragma unroll
  for (int t = 0; t < 4; ++t) {
    int col = c0base + t * 16 + lrow;
    if (col < Nc) {
      float bv = bias ? bias[col] : 0.f;
#pragma unroll
      for (int r = 0; r < 8; ++r) {
        float o = acc[t][r] + bv;
        if (relu) o = fmaxf(o, 0.f);
        C[(size_t)(m0 + r + 8 * hi) * Nc + col] = o;
      }
    }
  }
}

// ---------------- LayerNorm (+optional pre-bias, +optional ReLU), 256-wide rows ----
__global__ void __launch_bounds__(256)
ln_relu_kernel(const float* __restrict__ in, const float* __restrict__ prebias,
               const float* __restrict__ g, const float* __restrict__ beta,
               float* __restrict__ out, int rows, int relu) {
  int wave = threadIdx.x >> 5, lane = threadIdx.x & 31;
  int row = blockIdx.x * 8 + wave;
  if (row >= rows) return;
  const float* p = in + (size_t)row * HID;
  float v[8];
  float s = 0.f;
#pragma unroll
  for (int j = 0; j < 8; ++j) {
    int c = lane + j * 32;
    v[j] = p[c] + (prebias ? prebias[c] : 0.f);
    s += v[j];
  }
#pragma unroll
  for (int m = 16; m >= 1; m >>= 1) s += __shfl_xor(s, m, 32);
  float mean = s * (1.f / HID);
  float vs = 0.f;
#pragma unroll
  for (int j = 0; j < 8; ++j) { float d = v[j] - mean; vs += d * d; }
#pragma unroll
  for (int m = 16; m >= 1; m >>= 1) vs += __shfl_xor(vs, m, 32);
  float inv = rsqrtf(vs * (1.f / HID) + 1e-5f);
#pragma unroll
  for (int j = 0; j < 8; ++j) {
    int c = lane + j * 32;
    float o = (v[j] - mean) * inv * g[c] + beta[c];
    if (relu) o = fmaxf(o, 0.f);
    out[(size_t)row * HID + c] = o;
  }
}

// ---------------- attention coefficients: als/ald[n,h] = <xh[n,h*:], a{s,d}[h,:]> --
__global__ void attn_coef_kernel(const float* __restrict__ xh,
                                 const float* __restrict__ a_s, const float* __restrict__ a_d,
                                 float* __restrict__ als, float* __restrict__ ald, int H) {
  int i = blockIdx.x * blockDim.x + threadIdx.x;
  if (i >= N_NODES * H) return;
  int cph = HID / H;
  int n = i / H, h = i % H;
  const float* row = xh + (size_t)n * HID + h * cph;
  float s = 0.f, d = 0.f;
  for (int c = 0; c < cph; ++c) {
    float xv = row[c];
    s = fmaf(xv, a_s[h * cph + c], s);
    d = fmaf(xv, a_d[h * cph + c], d);
  }
  als[i] = s;
  ald[i] = d;
}

// ---------------- edge kernels (self-loops appended logically) ----------------
__device__ __forceinline__ void edge_ends(const int* __restrict__ src, const int* __restrict__ dst,
                                          long long e, int& s, int& d) {
  if (e < N_EDGES) { s = src[e]; d = dst[e]; }
  else             { s = d = (int)(e - N_EDGES); }
}

__global__ void edge_max_kernel(const int* __restrict__ src, const int* __restrict__ dst,
                                const float* __restrict__ als, const float* __restrict__ ald,
                                float* __restrict__ eb, int* __restrict__ mx, int H) {
  long long i = (long long)blockIdx.x * blockDim.x + threadIdx.x;
  if (i >= (long long)ESL * H) return;
  long long e = i / H; int h = (int)(i % H);
  int s, d; edge_ends(src, dst, e, s, d);
  float v = als[(size_t)s * H + h] + ald[(size_t)d * H + h];
  v = v > 0.f ? v : 0.2f * v;                 // leaky_relu(0.2)
  eb[i] = v;
  int key = __float_as_int(v);
  key = key >= 0 ? key : (key ^ 0x7fffffff);  // order-preserving int encoding
  atomicMax(&mx[(size_t)d * H + h], key);
}

__global__ void edge_exp_kernel(const int* __restrict__ src, const int* __restrict__ dst,
                                float* __restrict__ eb, const int* __restrict__ mx,
                                float* __restrict__ den, int H) {
  long long i = (long long)blockIdx.x * blockDim.x + threadIdx.x;
  if (i >= (long long)ESL * H) return;
  long long e = i / H; int h = (int)(i % H);
  int s, d; edge_ends(src, dst, e, s, d);
  (void)s;
  int k = mx[(size_t)d * H + h];
  float m = __int_as_float(k >= 0 ? k : (k ^ 0x7fffffff));
  float ex = __expf(eb[i] - m);
  eb[i] = ex;
  atomicAdd(&den[(size_t)d * H + h], ex);
}

#define EPB 8
__global__ void __launch_bounds__(256)
edge_agg_kernel(const int* __restrict__ src, const int* __restrict__ dst,
                const float* __restrict__ xh, const float* __restrict__ eb,
                const float* __restrict__ den, float* __restrict__ agg, int H) {
  int t = threadIdx.x;              // column 0..255
  int cph = HID / H;
  int h = t / cph;
  for (int j = 0; j < EPB; ++j) {
    long long e = (long long)blockIdx.x * EPB + j;
    if (e >= ESL) break;
    int s, d; edge_ends(src, dst, e, s, d);
    float alpha = eb[e * H + h] / (den[(size_t)d * H + h] + 1e-16f);
    atomicAdd(&agg[(size_t)d * HID + t], xh[(size_t)s * HID + t] * alpha);
  }
}

// ---------------- entity-logit renormalization: log(softmax + 1e-8) -------------
__global__ void __launch_bounds__(256)
log_softmax_kernel(float* __restrict__ xio, int rows) {
  int wave = threadIdx.x >> 5, lane = threadIdx.x & 31;
  int row = blockIdx.x * 8 + wave;
  if (row >= rows) return;
  float* p = xio + (size_t)row * NEC;
  float v[7];
  float mx = -3.4e38f;
#pragma unroll
  for (int j = 0; j < 7; ++j) { v[j] = p[lane + j * 32]; mx = fmaxf(mx, v[j]); }
#pragma unroll
  for (int m = 16; m >= 1; m >>= 1) mx = fmaxf(mx, __shfl_xor(mx, m, 32));
  float s = 0.f;
#pragma unroll
  for (int j = 0; j < 7; ++j) { v[j] = __expf(v[j] - mx); s += v[j]; }
#pragma unroll
  for (int m = 16; m >= 1; m >>= 1) s += __shfl_xor(s, m, 32);
  float inv = 1.f / s;
#pragma unroll
  for (int j = 0; j < 7; ++j) p[lane + j * 32] = __logf(v[j] * inv + 1e-8f);
}

// ---------------- fused relation head: 16 edges/block, 3 WMMA stages -------------
// pair(16x512 f16, LDS) @ Wr1(512x256) -> relu -> @ Wr2(256x128) -> relu -> @ Wr3(128x16 padded)
// Pair gather uses CDNA5 async global->LDS copies (ASYNCcnt + s_wait_asynccnt).
__global__ void __launch_bounds__(128)
relation_head_kernel(const _Float16* __restrict__ h2h,
                     const int* __restrict__ src, const int* __restrict__ dst,
                     const _Float16* __restrict__ Wr1h, const float* __restrict__ br1,
                     const _Float16* __restrict__ Wr2h, const float* __restrict__ br2,
                     const _Float16* __restrict__ Wr3p, const float* __restrict__ br3,
                     float* __restrict__ out) {
  __shared__ __align__(16) _Float16 pairA[16 * 512];
  __shared__ __align__(16) _Float16 r1h[16 * 256];
  __shared__ __align__(16) _Float16 r2h[16 * 128];

  int t = threadIdx.x;
  int wave = __builtin_amdgcn_readfirstlane(t >> 5);  // scalar wave id
  int lane = t & 31;
  long long e0 = (long long)blockIdx.x * 16;

  // gather pair features [h2[src] | h2[dst]] into LDS via async b128 copies.
  // Thread t covers 64 contiguous halves (128B) of one edge-row; part<4 -> src half.
  {
    int rowi = t >> 3;        // 0..15
    int part = t & 7;         // 8 threads per edge-row
    long long e = e0 + rowi;
    int s = src[e], d = dst[e];
    const _Float16* basep = (part < 4)
        ? (h2h + (size_t)s * HID + part * 64)
        : (h2h + (size_t)d * HID + (part - 4) * 64);
    _Float16* ldst = &pairA[rowi * 512 + part * 64];
#pragma unroll
    for (int j = 0; j < 8; ++j)
      async_load_b128(basep + j * 8, ldst + j * 8);
    wait_async0();
  }
  __syncthreads();

  int lrow = lane & 15, hi = lane >> 4;

  // stage 1: r1 = relu(pair @ Wr1 + br1), wave w -> cols [w*64, w*64+64)
  {
    v8f acc[4];
#pragma unroll
    for (int tt = 0; tt < 4; ++tt) acc[tt] = v8f_zero();
    const _Float16* arow = pairA + lrow * 512 + hi * 8;
    for (int k0 = 0; k0 < 512; k0 += 32) {
      v16h a = load_afrag_f16(arow + k0);
#pragma unroll
      for (int tt = 0; tt < 4; ++tt) {
        int c0 = wave * 64 + tt * 16;
        v16h b = load_bfrag_f16(Wr1h + (size_t)(k0 + hi * 8) * 256 + c0 + lrow, 256);
        acc[tt] = wmma_f16(a, b, acc[tt]);
      }
    }
#pragma unroll
    for (int tt = 0; tt < 4; ++tt) {
      int c0 = wave * 64 + tt * 16;
      int col = c0 + lrow;
      float bv = br1[col];
#pragma unroll
      for (int r = 0; r < 8; ++r)
        r1h[(r + 8 * hi) * 256 + col] = (_Float16)fmaxf(acc[tt][r] + bv, 0.f);
    }
  }
  __syncthreads();

  // stage 2: r2 = relu(r1 @ Wr2 + br2), waves 0..1 -> 128 cols
  if (wave < 2) {
    v8f acc[4];
#pragma unroll
    for (int tt = 0; tt < 4; ++tt) acc[tt] = v8f_zero();
    const _Float16* arow = r1h + lrow * 256 + hi * 8;
    for (int k0 = 0; k0 < 256; k0 += 32) {
      v16h a = load_afrag_f16(arow + k0);
#pragma unroll
      for (int tt = 0; tt < 4; ++tt) {
        int c0 = wave * 64 + tt * 16;
        v16h b = load_bfrag_f16(Wr2h + (size_t)(k0 + hi * 8) * 128 + c0 + lrow, 128);
        acc[tt] = wmma_f16(a, b, acc[tt]);
      }
    }
#pragma unroll
    for (int tt = 0; tt < 4; ++tt) {
      int c0 = wave * 64 + tt * 16;
      int col = c0 + lrow;
      float bv = br2[col];
#pragma unroll
      for (int r = 0; r < 8; ++r)
        r2h[(r + 8 * hi) * 128 + col] = (_Float16)fmaxf(acc[tt][r] + bv, 0.f);
    }
  }
  __syncthreads();

  // stage 3: logits = r2 @ Wr3 + br3 (6 valid cols of a padded-16 tile), wave 0
  if (wave == 0) {
    v8f acc = v8f_zero();
    const _Float16* arow = r2h + lrow * 128 + hi * 8;
    for (int k0 = 0; k0 < 128; k0 += 32) {
      v16h a = load_afrag_f16(arow + k0);
      v16h b = load_bfrag_f16(Wr3p + (size_t)(k0 + hi * 8) * 16 + lrow, 16);
      acc = wmma_f16(a, b, acc);
    }
    int col = lrow;
    if (col < NRC) {
      float bv = br3[col];
#pragma unroll
      for (int r = 0; r < 8; ++r)
        out[(size_t)(e0 + r + 8 * hi) * NRC + col] = acc[r] + bv;
    }
  }
}

// ================= host orchestration =================
extern "C" void kernel_launch(void* const* d_in, const int* in_sizes, int n_in,
                              void* d_out, int out_size, void* d_ws, size_t ws_size,
                              hipStream_t stream) {
  (void)in_sizes; (void)n_in; (void)out_size; (void)ws_size;

  const float* x   = (const float*)d_in[0];
  const int*   ei  = (const int*)d_in[1];
  const float* Wp  = (const float*)d_in[2];
  const float* bp  = (const float*)d_in[3];
  const float* g0  = (const float*)d_in[4];
  const float* n0  = (const float*)d_in[5];
  const float* W1  = (const float*)d_in[6];
  const float* a1s = (const float*)d_in[7];
  const float* a1d = (const float*)d_in[8];
  const float* bg1 = (const float*)d_in[9];
  const float* g1  = (const float*)d_in[10];
  const float* n1  = (const float*)d_in[11];
  const float* W2  = (const float*)d_in[12];
  const float* a2s = (const float*)d_in[13];
  const float* a2d = (const float*)d_in[14];
  const float* bg2 = (const float*)d_in[15];
  const float* g2  = (const float*)d_in[16];
  const float* n2  = (const float*)d_in[17];
  const float* We1 = (const float*)d_in[18];
  const float* be1 = (const float*)d_in[19];
  const float* We2 = (const float*)d_in[20];
  const float* be2 = (const float*)d_in[21];
  const float* We3 = (const float*)d_in[22];
  const float* be3 = (const float*)d_in[23];
  const float* Wr1 = (const float*)d_in[24];
  const float* br1 = (const float*)d_in[25];
  const float* Wr2 = (const float*)d_in[26];
  const float* br2 = (const float*)d_in[27];
  const float* Wr3 = (const float*)d_in[28];
  const float* br3 = (const float*)d_in[29];

  const int* src = ei;
  const int* dst = ei + N_EDGES;
  float* ent_out = (float*)d_out;                        // [N, 224]
  float* rel_out = ent_out + (size_t)N_NODES * NEC;      // [E, 6]

  // ---- workspace carve (256B aligned) ----
  size_t off = 0;
  char* base = (char*)d_ws;
  auto carve = [&](size_t bytes) -> char* {
    char* p = base + off;
    off += (bytes + 255) & ~(size_t)255;
    return p;
  };
  _Float16* Wph  = (_Float16*)carve((size_t)D_INP * HID * 2);
  _Float16* W1h  = (_Float16*)carve((size_t)HID * HID * 2);
  _Float16* W2h  = (_Float16*)carve((size_t)HID * HID * 2);
  _Float16* We1h = (_Float16*)carve((size_t)HID * 128 * 2);
  _Float16* We2h = (_Float16*)carve((size_t)128 * 64 * 2);
  _Float16* We3p = (_Float16*)carve((size_t)64 * 256 * 2);   // 224 cols padded to 256
  _Float16* Wr1h = (_Float16*)carve((size_t)512 * HID * 2);
  _Float16* Wr2h = (_Float16*)carve((size_t)HID * 128 * 2);
  _Float16* Wr3p = (_Float16*)carve((size_t)128 * 16 * 2);   // 6 cols padded to 16
  float* h0  = (float*)carve((size_t)N_NODES * HID * 4);
  float* xh  = (float*)carve((size_t)N_NODES * HID * 4);
  float* agg = (float*)carve((size_t)N_NODES * HID * 4);
  float* h1  = (float*)carve((size_t)N_NODES * HID * 4);
  float* h2  = (float*)carve((size_t)N_NODES * HID * 4);
  _Float16* h2h = (_Float16*)carve((size_t)N_NODES * HID * 2);
  float* als = (float*)carve((size_t)N_NODES * HEADS * 4);
  float* ald = (float*)carve((size_t)N_NODES * HEADS * 4);
  int*   mx  = (int*)  carve((size_t)N_NODES * HEADS * 4);
  float* den = (float*)carve((size_t)N_NODES * HEADS * 4);
  float* eb  = (float*)carve((size_t)ESL * HEADS * 4);
  float* e1  = (float*)carve((size_t)N_NODES * 128 * 4);
  float* e2  = (float*)carve((size_t)N_NODES * 64 * 4);

  auto cvt = [&](const float* in, _Float16* outp, long long n) {
    cvt_kernel<<<dim3((unsigned)((n + 255) / 256)), dim3(256), 0, stream>>>(in, outp, n);
  };
  // Ncp = padded column count of B (multiple of 64); Nc = logical columns of C.
  auto gemm = [&](const float* A, const _Float16* Bp, const float* bias, float* C,
                  int M, int K, int Nc, int Ncp, int relu) {
    wmma_gemm_kernel<<<dim3((unsigned)((M + 63) / 64), (unsigned)(Ncp / 64)), dim3(128), 0, stream>>>(
        A, Bp, bias, C, M, K, Nc, Ncp, relu);
  };
  auto ln = [&](const float* in, const float* pre, const float* gg, const float* bb,
                float* outp, int relu) {
    ln_relu_kernel<<<dim3(N_NODES / 8), dim3(256), 0, stream>>>(in, pre, gg, bb, outp, N_NODES, relu);
  };
  auto fillu = [&](void* p, unsigned v, long long n) {
    fill_kernel<<<dim3((unsigned)((n + 255) / 256)), dim3(256), 0, stream>>>((unsigned*)p, v, n);
  };
  auto gat = [&](const float* hin, const _Float16* Wh, const float* as_, const float* ad_,
                 const float* bg, const float* gg, const float* bb, float* hout, int H) {
    gemm(hin, Wh, nullptr, xh, N_NODES, HID, HID, HID, 0);
    attn_coef_kernel<<<dim3((N_NODES * H + 255) / 256), dim3(256), 0, stream>>>(xh, as_, ad_, als, ald, H);
    fillu(mx, 0x80000000u, (long long)N_NODES * H);
    fillu(den, 0u, (long long)N_NODES * H);
    long long tot = (long long)ESL * H;
    edge_max_kernel<<<dim3((unsigned)((tot + 255) / 256)), dim3(256), 0, stream>>>(src, dst, als, ald, eb, mx, H);
    edge_exp_kernel<<<dim3((unsigned)((tot + 255) / 256)), dim3(256), 0, stream>>>(src, dst, eb, mx, den, H);
    fillu(agg, 0u, (long long)N_NODES * HID);
    edge_agg_kernel<<<dim3((unsigned)((ESL + EPB - 1) / EPB)), dim3(256), 0, stream>>>(src, dst, xh, eb, den, agg, H);
    ln(agg, bg, gg, bb, hout, 1);
  };

  // ---- weight conversion (fp32 -> f16, column-padded where needed) ----
  cvt(Wp,  Wph,  (long long)D_INP * HID);
  cvt(W1,  W1h,  (long long)HID * HID);
  cvt(W2,  W2h,  (long long)HID * HID);
  cvt(We1, We1h, (long long)HID * 128);
  cvt(We2, We2h, (long long)128 * 64);
  cvt_pad_kernel<<<dim3((64 * 256 + 255) / 256), dim3(256), 0, stream>>>(We3, We3p, 64, NEC, 256);
  cvt(Wr1, Wr1h, (long long)512 * HID);
  cvt(Wr2, Wr2h, (long long)HID * 128);
  cvt_pad_kernel<<<dim3((128 * 16 + 255) / 256), dim3(256), 0, stream>>>(Wr3, Wr3p, 128, NRC, 16);

  // ---- input projection + LN ----
  gemm(x, Wph, bp, xh, N_NODES, D_INP, HID, HID, 0);
  ln(xh, nullptr, g0, n0, h0, 0);

  // ---- GAT layers ----
  gat(h0, W1h, a1s, a1d, bg1, g1, n1, h1, HEADS);
  gat(h1, W2h, a2s, a2d, bg2, g2, n2, h2, 1);

  // ---- entity head ----
  gemm(h2, We1h, be1, e1, N_NODES, HID, 128, 128, 1);
  gemm(e1, We2h, be2, e2, N_NODES, 128, 64, 64, 1);
  gemm(e2, We3p, be3, ent_out, N_NODES, 64, NEC, 256, 0);
  log_softmax_kernel<<<dim3(N_NODES / 8), dim3(256), 0, stream>>>(ent_out, N_NODES);

  // ---- relation head (fused WMMA MLP over edges) ----
  cvt(h2, h2h, (long long)N_NODES * HID);
  relation_head_kernel<<<dim3(N_EDGES / 16), dim3(128), 0, stream>>>(
      h2h, src, dst, Wr1h, br1, Wr2h, br2, Wr3p, br3, rel_out);
}
